// MultiHeadSelfAttentionLayer_58523224375379
// MI455X (gfx1250) — matrix operand order repaired and checked by reading, not compiled, not run on hardware
//
#include <hip/hip_runtime.h>

// ---------- types ----------
typedef __attribute__((ext_vector_type(16))) __bf16        v16bf;
typedef __attribute__((ext_vector_type(8)))  float         v8f;
typedef __attribute__((ext_vector_type(4)))  float         v4f;
typedef __attribute__((ext_vector_type(4)))  unsigned int  v4u;
typedef unsigned short u16;
typedef unsigned int   u32;

struct BFr { v4u a, b; };   // 32B == v16bf

// ---------- helpers ----------
static __device__ __forceinline__ u16 bf16rne(float f) {
  u32 u = __float_as_uint(f);
  u += 0x7fffu + ((u >> 16) & 1u);   // round-to-nearest-even to bf16
  return (u16)(u >> 16);
}
static __device__ __forceinline__ float bf16tof(u16 h) {
  return __uint_as_float(((u32)h) << 16);
}
static __device__ __forceinline__ v16bf mkfrag(v4u a, v4u b) {
  BFr t{a, b};
  return __builtin_bit_cast(v16bf, t);
}
static __device__ __forceinline__ v8f wmma_bf16(v16bf a, v16bf b, v8f c) {
  // v_wmma_f32_16x16x32_bf16  D = A*B + C
  return __builtin_amdgcn_wmma_f32_16x16x32_bf16(false, a, false, b, (short)0, c,
                                                 false, false);
}
// split 8 f32 (K-ascending) into bf16 hi / bf16 lo packed pairs
static __device__ __forceinline__ void split8(v4f a, v4f b, v4u& hi, v4u& lo) {
  float f[8] = {a.x, a.y, a.z, a.w, b.x, b.y, b.z, b.w};
  u32 h[8], l[8];
#pragma unroll
  for (int i = 0; i < 8; i++) {
    u16 hh = bf16rne(f[i]);
    h[i] = hh;
    l[i] = bf16rne(f[i] - bf16tof(hh));
  }
  hi.x = h[0] | (h[1] << 16); hi.y = h[2] | (h[3] << 16);
  hi.z = h[4] | (h[5] << 16); hi.w = h[6] | (h[7] << 16);
  lo.x = l[0] | (l[1] << 16); lo.y = l[2] | (l[3] << 16);
  lo.z = l[4] | (l[5] << 16); lo.w = l[6] | (l[7] << 16);
}

// ---------- kernel 1: weight f32 -> transposed bf16 hi/lo planes ----------
// W (1024 k x 1024 n) f32  ->  Wt (1024 n x 1024 k) bf16 hi plane + lo plane
__global__ __launch_bounds__(256) void conv_weights(
    const float* __restrict__ W0, const float* __restrict__ W1,
    const float* __restrict__ W2, const float* __restrict__ W3,
    u16* __restrict__ Wt) {
  __shared__ u16 sh[64 * 65];
  __shared__ u16 sl[64 * 65];
  int w    = blockIdx.x >> 8;
  int tile = blockIdx.x & 255;
  int k0 = (tile >> 4) * 64;
  int n0 = (tile & 15) * 64;
  const float* W = (w == 0) ? W0 : (w == 1) ? W1 : (w == 2) ? W2 : W3;
  u16* oh = Wt + (size_t)w * 2097152u;   // 1M hi + 1M lo per weight
  u16* ol = oh + 1048576u;
#pragma unroll
  for (int i = 0; i < 16; i++) {
    int e = threadIdx.x + i * 256;
    int rk = e >> 6, cn = e & 63;
    float f = W[(size_t)(k0 + rk) * 1024 + (n0 + cn)];
    u16 h = bf16rne(f);
    sh[cn * 65 + rk] = h;
    sl[cn * 65 + rk] = bf16rne(f - bf16tof(h));
  }
  __syncthreads();
#pragma unroll
  for (int i = 0; i < 16; i++) {
    int e = threadIdx.x + i * 256;
    int rn = e >> 6, ck = e & 63;
    size_t off = (size_t)(n0 + rn) * 1024 + (k0 + ck);
    oh[off] = sh[rn * 65 + ck];
    ol[off] = sl[rn * 65 + ck];
  }
}

// ---------- kernel 2: elementwise f32 -> bf16 hi/lo planes (for X) ----------
// exactly 8 elements per thread; grid sized so coverage is exact
__global__ __launch_bounds__(256) void conv_f32_bf16x2(
    const float* __restrict__ A, u16* __restrict__ hi, u16* __restrict__ lo) {
  int i = blockIdx.x * 256 + threadIdx.x;   // 8-element group index
  const v4f* a4 = (const v4f*)A;
  v4f t0 = a4[2 * i], t1 = a4[2 * i + 1];
  v4u hv, lv;
  split8(t0, t1, hv, lv);
  ((v4u*)hi)[i] = hv;
  ((v4u*)lo)[i] = lv;
}

// ---------- kernel 3: split-bf16 GEMM  C(8192x1024) = A * W + bias ----------
// A pre-split into bf16 hi/lo planes. WG = 8 waves; wave -> 16 rows x 64 cols;
// B staged in LDS (shared by all 8 waves).
// mode 0: write f32 to oF.  mode 1: write scaled bf16 hi/lo planes (Q/K).
__global__ __launch_bounds__(256) void gemm_bb(
    const u16* __restrict__ Ahi, const u16* __restrict__ Alo,
    const u16* __restrict__ BtHi, const u16* __restrict__ BtLo,
    const float* __restrict__ bias, float scale, int mode,
    u16* __restrict__ oHi, u16* __restrict__ oLo, float* __restrict__ oF) {
  __shared__ __align__(16) u16 lb[2][2048];   // [hi|lo][64 n-rows x 32 k]
  int wave = threadIdx.x >> 5;
  int lane = threadIdx.x & 31;
  int ln = lane & 15, hf = lane >> 4;
  int m0 = blockIdx.x * 128 + wave * 16;
  int n0 = blockIdx.y * 64;
  v8f acc[4];
#pragma unroll
  for (int t = 0; t < 4; t++)
#pragma unroll
    for (int i = 0; i < 8; i++) acc[t][i] = 0.f;

  for (int k0 = 0; k0 < 1024; k0 += 32) {
    __syncthreads();
#pragma unroll
    for (int i = 0; i < 2; i++) {          // stage 8KB of B (hi+lo)
      int idx = threadIdx.x + i * 256;
      int pl = idx >> 8, j = idx & 255;
      int r = j >> 2, q = j & 3;
      const u16* src = (pl ? BtLo : BtHi) + (size_t)(n0 + r) * 1024 + k0;
      ((v4u*)lb[pl])[r * 4 + q] = ((const v4u*)src)[q];
    }
    __syncthreads();
    // A fragment: lane row m0+ln; A-layout K = {8h..8h+7} u {16+8h..23+8h}
    const v4u* xh = (const v4u*)(Ahi + (size_t)(m0 + ln) * 1024 + k0);
    const v4u* xl = (const v4u*)(Alo + (size_t)(m0 + ln) * 1024 + k0);
    v16bf aH = mkfrag(xh[hf], xh[2 + hf]);
    v16bf aL = mkfrag(xl[hf], xl[2 + hf]);
    const v4u* bh = (const v4u*)lb[0];
    const v4u* bl = (const v4u*)lb[1];
#pragma unroll
    for (int t = 0; t < 4; t++) {
      // B fragment: lane col t*16+ln; B-layout K = {8h..8h+7} u {16+8h..23+8h}
      int base = (t * 16 + ln) * 4 + hf;
      v16bf bH = mkfrag(bh[base], bh[base + 2]);
      v16bf bL = mkfrag(bl[base], bl[base + 2]);
      acc[t] = wmma_bf16(aH, bH, acc[t]);
      acc[t] = wmma_bf16(aH, bL, acc[t]);
      acc[t] = wmma_bf16(aL, bH, acc[t]);
    }
  }
  // epilogue: C layout -> row = m0 + i + 8*hf, col = n0 + t*16 + ln
#pragma unroll
  for (int t = 0; t < 4; t++) {
    int col = n0 + t * 16 + ln;
    float bv = bias[col];
#pragma unroll
    for (int i = 0; i < 8; i++) {
      int row = m0 + i + 8 * hf;
      float v = (acc[t][i] + bv) * scale;
      size_t off = (size_t)row * 1024 + col;
      if (mode == 0) {
        oF[off] = v;
      } else {
        u16 h = bf16rne(v);
        oHi[off] = h;
        oLo[off] = bf16rne(v - bf16tof(h));
      }
    }
  }
}

// ---------- kernel 4: per-head diagonal softmax attention ----------
// Head (n,h) is a contiguous (2048 x 64) block. WG: one head, 128 columns.
// Wave: 16 columns b; K-fragments resident; stream Q in 16-row chunks via LDS.
// Column-softmax stats (over query axis a) reduced online; diag tile captured.
// Output written as bf16 hi/lo planes for the final GEMM.
__global__ __launch_bounds__(256) void attn_diag(
    const u16* __restrict__ Qhi, const u16* __restrict__ Qlo,
    const u16* __restrict__ Khi, const u16* __restrict__ Klo,
    const float* __restrict__ V,
    u16* __restrict__ Ohi, u16* __restrict__ Olo) {
  __shared__ __align__(16) u16 sA[2][1024];   // 16 rows x 64 x, hi+lo
  int nh = blockIdx.y;
  size_t hb = (size_t)nh * 131072u;           // head base (elements)
  int wave = threadIdx.x >> 5;
  int lane = threadIdx.x & 31;
  int ln = lane & 15, hf = lane >> 4;
  int c0 = blockIdx.x * 128 + wave * 16;
  // resident B (= K rows for this wave's 16 columns), K-dim = 64 -> 2 ksteps
  size_t kb = hb + (size_t)(c0 + ln) * 64;
  const v4u* kh = (const v4u*)(Khi + kb);
  const v4u* kl = (const v4u*)(Klo + kb);
  v16bf b0h = mkfrag(kh[hf],     kh[2 + hf]);
  v16bf b0l = mkfrag(kl[hf],     kl[2 + hf]);
  v16bf b1h = mkfrag(kh[4 + hf], kh[6 + hf]);
  v16bf b1l = mkfrag(kl[4 + hf], kl[6 + hf]);

  float m_run = -1e30f, s_run = 0.f, diag = 0.f;
  for (int a0 = 0; a0 < 2048; a0 += 16) {
    __syncthreads();
    {   // stage 16 Q rows (contiguous 2KB per plane)
      int t = threadIdx.x;
      int pl = t >> 7, j = t & 127;
      const u16* src = (pl ? Qlo : Qhi) + hb + (size_t)a0 * 64;
      ((v4u*)sA[pl])[j] = ((const v4u*)src)[j];
    }
    __syncthreads();
    const v4u* ah = (const v4u*)sA[0];
    const v4u* al = (const v4u*)sA[1];
    int rb = ln * 8;
    v16bf a0h = mkfrag(ah[rb + hf],     ah[rb + 2 + hf]);
    v16bf a0l = mkfrag(al[rb + hf],     al[rb + 2 + hf]);
    v16bf a1h = mkfrag(ah[rb + 4 + hf], ah[rb + 6 + hf]);
    v16bf a1l = mkfrag(al[rb + 4 + hf], al[rb + 6 + hf]);
    v8f c;
#pragma unroll
    for (int i = 0; i < 8; i++) c[i] = 0.f;
    c = wmma_bf16(a0h, b0h, c);
    c = wmma_bf16(a0h, b0l, c);
    c = wmma_bf16(a0l, b0h, c);
    c = wmma_bf16(a1h, b1h, c);
    c = wmma_bf16(a1h, b1l, c);
    c = wmma_bf16(a1l, b1h, c);
    // online column softmax stats (reduce over M: 8 in-lane + xor-16)
    float mb = c[0];
#pragma unroll
    for (int i = 1; i < 8; i++) mb = fmaxf(mb, c[i]);
    mb = fmaxf(mb, __shfl_xor(mb, 16));
    float mn = fmaxf(m_run, mb);
    float p = 0.f;
#pragma unroll
    for (int i = 0; i < 8; i++) p += __expf(c[i] - mn);
    p += __shfl_xor(p, 16);
    s_run = s_run * __expf(m_run - mn) + p;
    m_run = mn;
    if (a0 == c0) {                 // tile on the global diagonal
      int sel = ln & 7;
      float dv = c[0];
#pragma unroll
      for (int i = 1; i < 8; i++)
        if (sel == i) dv = c[i];
      float other = __shfl_xor(dv, 16);
      bool valid = ((hf == 1) == (ln >= 8));   // M==N lives in matching half
      diag = valid ? dv : other;
    }
  }
  float sc = __expf(diag - m_run) / s_run;     // attn[b,b]
  // out[b,:] = sc * V[b,:]; lane writes 32 elements, pre-split to bf16 hi/lo
  size_t vb = hb + (size_t)(c0 + ln) * 64 + (size_t)hf * 32;
  const v4f* vp = (const v4f*)(V + vb);
  v4u* oph = (v4u*)(Ohi + vb);
  v4u* opl = (v4u*)(Olo + vb);
#pragma unroll
  for (int q = 0; q < 4; q++) {
    v4f t0 = vp[2 * q] * sc;
    v4f t1 = vp[2 * q + 1] * sc;
    v4u hv, lv;
    split8(t0, t1, hv, lv);
    oph[q] = hv;
    opl[q] = lv;
  }
}

// ---------- host ----------
extern "C" void kernel_launch(void* const* d_in, const int* in_sizes, int n_in,
                              void* d_out, int out_size, void* d_ws, size_t ws_size,
                              hipStream_t stream) {
  (void)in_sizes; (void)n_in; (void)out_size; (void)ws_size;
  const float* X  = (const float*)d_in[0];
  const float* Wq = (const float*)d_in[1];
  const float* bq = (const float*)d_in[2];
  const float* Wk = (const float*)d_in[3];
  const float* bk = (const float*)d_in[4];
  const float* Wv = (const float*)d_in[5];
  const float* bv = (const float*)d_in[6];
  const float* Wo = (const float*)d_in[7];
  const float* bo = (const float*)d_in[8];
  float* out = (float*)d_out;
  char* ws = (char*)d_ws;
  const size_t MB = 1024u * 1024u;
  // workspace layout (176 MB total)
  u16*   Wt  = (u16*)(ws);               // 16 MB: 4 x (hi 2MB + lo 2MB), transposed
  u16*   Xhi = (u16*)(ws + 16 * MB);     // 16 MB per plane from here on
  u16*   Xlo = (u16*)(ws + 32 * MB);
  u16*   Qhi = (u16*)(ws + 48 * MB);
  u16*   Qlo = (u16*)(ws + 64 * MB);
  u16*   Khi = (u16*)(ws + 80 * MB);
  u16*   Klo = (u16*)(ws + 96 * MB);
  float* Vf  = (float*)(ws + 112 * MB);  // 32 MB
  u16*   Ohi = (u16*)(ws + 144 * MB);
  u16*   Olo = (u16*)(ws + 160 * MB);

  conv_weights<<<1024, 256, 0, stream>>>(Wq, Wk, Wv, Wo, Wt);
  conv_f32_bf16x2<<<4096, 256, 0, stream>>>(X, Xhi, Xlo);   // 8M elems / 8 per thread

  const u16* Wtq = Wt;
  const u16* Wtk = Wt + 2097152u;
  const u16* Wtv = Wt + 4194304u;
  const u16* Wto = Wt + 6291456u;
  const float kScale = 0.03125f;        // 1/sqrt(1024): applied to Q and K each
  dim3 g(64, 16);
  gemm_bb<<<g, 256, 0, stream>>>(Xhi, Xlo, Wtq, Wtq + 1048576u, bq, kScale, 1,
                                 Qhi, Qlo, nullptr);
  gemm_bb<<<g, 256, 0, stream>>>(Xhi, Xlo, Wtk, Wtk + 1048576u, bk, kScale, 1,
                                 Khi, Klo, nullptr);
  gemm_bb<<<g, 256, 0, stream>>>(Xhi, Xlo, Wtv, Wtv + 1048576u, bv, 1.0f, 0,
                                 nullptr, nullptr, Vf);

  attn_diag<<<dim3(16, 64), 256, 0, stream>>>(Qhi, Qlo, Khi, Klo, Vf, Ohi, Olo);

  gemm_bb<<<g, 256, 0, stream>>>(Ohi, Olo, Wto, Wto + 1048576u, bo, 1.0f, 0,
                                 nullptr, nullptr, out);
}
